// TransformerBlock_40793599377662
// MI455X (gfx1250) — matrix operand order repaired and checked by reading, compiled, not run on hardware
//
#include <hip/hip_runtime.h>
#include <hip/hip_bf16.h>
#include <cstddef>

// Problem constants (match reference)
#define BB 2
#define SS 2048
#define DD 1024
#define HH 16
#define DHH 64
#define FFF 4096
#define MROWS (BB * SS)

typedef __attribute__((ext_vector_type(16))) __bf16 v16bf;
typedef __attribute__((ext_vector_type(8)))  float  v8f;
typedef __attribute__((ext_vector_type(4)))  unsigned int u32x4;
typedef __attribute__((ext_vector_type(4)))  int i32x4;
typedef __attribute__((ext_vector_type(8)))  int i32x8;

#if defined(__gfx1250__) && __has_builtin(__builtin_amdgcn_tensor_load_to_lds)
#define HAVE_TDM 1
#else
#define HAVE_TDM 0
#endif

union Frag { v16bf v; uint4 q[2]; };

__device__ __forceinline__ unsigned short f32_to_bf16(float f) {
  unsigned int u = __float_as_uint(f);
  unsigned int r = u + 0x7FFFu + ((u >> 16) & 1u);   // round-to-nearest-even
  return (unsigned short)(r >> 16);
}
__device__ __forceinline__ float bf16_to_f32(unsigned short h) {
  return __uint_as_float(((unsigned int)h) << 16);
}

__device__ __forceinline__ float hmax16(float v) {
  #pragma unroll
  for (int m = 1; m <= 8; m <<= 1) v = fmaxf(v, __shfl_xor(v, m, 32));
  return v;
}
__device__ __forceinline__ float hsum16(float v) {
  #pragma unroll
  for (int m = 1; m <= 8; m <<= 1) v += __shfl_xor(v, m, 32);
  return v;
}

#if HAVE_TDM
// 2D TDM load: tile_rows x tile_cols of bf16 (2B) from row-major tensor with
// row stride `stride_elems` into LDS at byte offset lds_off.
// D# layout per CDNA5 ISA.
__device__ __forceinline__ void tdm_load_2d(const unsigned short* gsrc, unsigned lds_off,
                                            unsigned tile_cols, unsigned tile_rows,
                                            unsigned tensor_cols, unsigned tensor_rows,
                                            unsigned stride_elems) {
  unsigned long long ga = (unsigned long long)(size_t)gsrc;
  u32x4 g0 = { 1u,                                   // count=1, user descriptor
               lds_off,                              // lds_addr (bytes)
               (unsigned)(ga & 0xFFFFFFFFu),         // global_addr[31:0]
               (unsigned)((ga >> 32) & 0x01FFFFFFu) | (2u << 30) }; // addr[56:32] | type=2
  i32x8 g1 = {
    (int)0x00010000u,                                            // data_size=1 (2 bytes)
    (int)((tensor_cols & 0xFFFFu) << 16),                        // tensor_dim0[15:0]
    (int)(((tensor_cols >> 16) & 0xFFFFu) |
          ((tensor_rows & 0xFFFFu) << 16)),                      // dim0 hi | dim1 lo
    (int)(((tensor_rows >> 16) & 0xFFFFu) |
          ((tile_cols & 0xFFFFu) << 16)),                        // dim1 hi | tile_dim0
    (int)(tile_rows & 0xFFFFu),                                  // tile_dim1 (tile_dim2=0)
    (int)stride_elems,                                           // tensor_dim0_stride lo
    0, 0 };
  i32x4 z4 = { 0, 0, 0, 0 };
#if __clang_major__ >= 23
  i32x8 z8 = { 0, 0, 0, 0, 0, 0, 0, 0 };
  __builtin_amdgcn_tensor_load_to_lds(g0, g1, z4, z4, z8, 0);
#else
  __builtin_amdgcn_tensor_load_to_lds(g0, g1, z4, z4, 0);
#endif
}
#endif

// ---------------------------------------------------------------- convert
__global__ __launch_bounds__(256)
void cvt_f32_bf16(const float* __restrict__ in, unsigned short* __restrict__ out, size_t n) {
  size_t i = (size_t)blockIdx.x * blockDim.x + threadIdx.x;
  if (i < n) out[i] = f32_to_bf16(in[i]);
}

// ---------------------------------------------------------------- RMSNorm
__global__ __launch_bounds__(256)
void rmsnorm_kernel(const float* __restrict__ x, const float* __restrict__ g,
                    unsigned short* __restrict__ out, int Dn) {
  __shared__ float red[256];
  int row = blockIdx.x;
  const float* xr = x + (size_t)row * Dn;
  float ss = 0.f;
  for (int d = threadIdx.x; d < Dn; d += 256) { float v = xr[d]; ss += v * v; }
  red[threadIdx.x] = ss;
  __syncthreads();
  #pragma unroll
  for (int s = 128; s > 0; s >>= 1) {
    if ((int)threadIdx.x < s) red[threadIdx.x] += red[threadIdx.x + s];
    __syncthreads();
  }
  float scale = rsqrtf(red[0] / (float)Dn + 1e-6f);
  for (int d = threadIdx.x; d < Dn; d += 256)
    out[(size_t)row * Dn + d] = f32_to_bf16(g[d] * xr[d] * scale);
}

// ---------------------------------------------------------------- RoPE (in-place, bf16)
__global__ __launch_bounds__(256)
void rope_kernel(unsigned short* __restrict__ qk) {
  size_t idx = (size_t)blockIdx.x * blockDim.x + threadIdx.x; // B*S*H*32 threads
  int j = (int)(idx & 31);
  size_t rest = idx >> 5;
  int h = (int)(rest % HH); rest /= HH;
  int s = (int)(rest % SS);
  int b = (int)(rest / SS);
  float inv = powf(10000.f, -(float)(2 * j) / 64.f);
  float ang = (float)s * inv;
  float c = __cosf(ang), sn = __sinf(ang);
  size_t base = (((size_t)(b * SS + s)) * HH + h) * DHH;
  float x1 = bf16_to_f32(qk[base + j]);
  float x2 = bf16_to_f32(qk[base + j + 32]);
  qk[base + j]      = f32_to_bf16(x1 * c - x2 * sn);
  qk[base + j + 32] = f32_to_bf16(x2 * c + x1 * sn);
}

// ---------------------------------------------------------------- bf16 GEMM 128x128 tile, 8 waves
// C = A(MxK) * B(KxN);  EP: 0 -> bf16 C
//                       1 -> f32 C = resid + C
//                       2 -> bf16 C = silu(aux) * C   (SwiGLU combine)
// A tile moved by the Tensor Data Mover (tensor_load_to_lds + s_wait_tensorcnt),
// B tile register-staged + transposed into LDS so B fragments are contiguous.
template <int EP>
__global__ __launch_bounds__(256)
void gemm_bf16_128x128(const unsigned short* __restrict__ A,
                       const unsigned short* __restrict__ Bm,
                       void* __restrict__ Cout,
                       const float* __restrict__ resid,
                       const unsigned short* __restrict__ aux,
                       int M, int N, int K) {
  __shared__ unsigned short As[128 * 64];   // [row][k]     16 KB
  __shared__ unsigned short Bs[128 * 64];   // [n][k] (T)   16 KB
  const int tid  = threadIdx.x;
  const int bm   = blockIdx.y * 128;
  const int bn   = blockIdx.x * 128;
  const int w    = tid >> 5;
  const int lane = tid & 31;
  const int lh   = lane & 15;
  const int hb   = lane >> 4;
  const int wm   = (w >> 1) * 32;
  const int wn   = (w & 1) * 64;

  v8f zero = {};
  v8f acc[2][4];
  #pragma unroll
  for (int i = 0; i < 2; ++i)
    #pragma unroll
    for (int j = 0; j < 4; ++j) acc[i][j] = zero;

  const int bkr  = tid >> 2;            // B loader: k row 0..63
  const int bnof = (tid & 3) * 32;      // B loader: n offset {0,32,64,96}
#if !HAVE_TDM
  const int ar = tid >> 1, ak = (tid & 1) * 32;   // A loader fallback
#endif

  for (int k0 = 0; k0 < K; k0 += 64) {
    // B tile: global -> regs (coalesced 64B per thread)
    const uint4* gB = (const uint4*)(Bm + (size_t)(k0 + bkr) * N + bn + bnof);
    uint4 b0 = gB[0], b1 = gB[1], b2 = gB[2], b3 = gB[3];
#if !HAVE_TDM
    const uint4* gA = (const uint4*)(A + (size_t)(bm + ar) * K + k0 + ak);
    uint4 a0 = gA[0], a1 = gA[1], a2 = gA[2], a3 = gA[3];
#endif
    if (k0 + 64 < K)
      __builtin_prefetch(Bm + (size_t)(k0 + 64 + bkr) * N + bn + bnof, 0, 1);

    __syncthreads();                    // previous iteration finished reading LDS
#if HAVE_TDM
    if (w == 0)                         // one TDM descriptor moves the whole A tile
      tdm_load_2d(A + (size_t)bm * K + k0,
                  (unsigned)(size_t)(void*)As,
                  /*tile_cols*/64, /*tile_rows*/128,
                  /*tensor_cols*/(unsigned)(K - k0), /*tensor_rows*/(unsigned)(M - bm),
                  /*stride*/(unsigned)K);
#else
    *(uint4*)(As + ar * 64 + ak)      = a0;
    *(uint4*)(As + ar * 64 + ak + 8)  = a1;
    *(uint4*)(As + ar * 64 + ak + 16) = a2;
    *(uint4*)(As + ar * 64 + ak + 24) = a3;
#endif
    unsigned short tb[32];
    *(uint4*)(tb)      = b0;
    *(uint4*)(tb + 8)  = b1;
    *(uint4*)(tb + 16) = b2;
    *(uint4*)(tb + 24) = b3;
    #pragma unroll
    for (int j = 0; j < 32; ++j) Bs[(bnof + j) * 64 + bkr] = tb[j];
#if HAVE_TDM
    if (w == 0) __builtin_amdgcn_s_wait_tensorcnt(0);   // TENSORcnt is per-wave
#endif
    __syncthreads();

    // two K=32 chunks per LDS tile
    #pragma unroll
    for (int c = 0; c < 2; ++c) {
      Frag af[2];
      #pragma unroll
      for (int i = 0; i < 2; ++i) {
        int row = wm + i * 16 + lh;
        af[i].q[0] = *(const uint4*)(As + row * 64 + c * 32 + hb * 8);
        af[i].q[1] = *(const uint4*)(As + row * 64 + c * 32 + hb * 8 + 16);
      }
      Frag bf[4];
      #pragma unroll
      for (int j = 0; j < 4; ++j) {
        int col = wn + j * 16 + lh;
        bf[j].q[0] = *(const uint4*)(Bs + col * 64 + c * 32 + hb * 8);
        bf[j].q[1] = *(const uint4*)(Bs + col * 64 + c * 32 + hb * 8 + 16);
      }
      #pragma unroll
      for (int i = 0; i < 2; ++i)
        #pragma unroll
        for (int j = 0; j < 4; ++j)
          acc[i][j] = __builtin_amdgcn_wmma_f32_16x16x32_bf16(
              false, af[i].v, false, bf[j].v, (short)0, acc[i][j], false, false);
    }
  }

  // straight-line epilogue (EP is compile-time)
  #pragma unroll
  for (int i = 0; i < 2; ++i)
    #pragma unroll
    for (int j = 0; j < 4; ++j) {
      int r0 = bm + wm + i * 16 + hb * 8;
      int c  = bn + wn + j * 16 + lh;
      #pragma unroll
      for (int r = 0; r < 8; ++r) {
        size_t idx = (size_t)(r0 + r) * N + c;
        float val = acc[i][j][r];
        if (EP == 0) {
          ((unsigned short*)Cout)[idx] = f32_to_bf16(val);
        } else if (EP == 1) {
          ((float*)Cout)[idx] = resid[idx] + val;
        } else {
          float a = bf16_to_f32(aux[idx]);
          float s = a / (1.0f + __expf(-a));
          ((unsigned short*)Cout)[idx] = f32_to_bf16(s * val);
        }
      }
    }
}

// ---------------------------------------------------------------- flash attention (bf16 WMMA)
// grid: (S/64, H, B); block: 128 (4 waves, 16 query rows each)
__global__ __launch_bounds__(128)
void flash_attn_kernel(const unsigned short* __restrict__ Q,
                       const unsigned short* __restrict__ Kp,
                       const unsigned short* __restrict__ V,
                       unsigned short* __restrict__ Op) {
  __shared__ unsigned short Qs[64 * 64];       // [q][dh]
  __shared__ unsigned short Ks[64 * 64];       // [key][dh]
  __shared__ unsigned short Vt[64 * 64];       // [dh][key] (transposed)
  __shared__ unsigned short Ps[4 * 16 * 64];   // per-wave P tile [q16][key64]

  const int b  = blockIdx.z, h = blockIdx.y, q0 = blockIdx.x * 64;
  const int tid  = threadIdx.x;
  const int w    = tid >> 5;
  const int lane = tid & 31;
  const int lh   = lane & 15;
  const int hb   = lane >> 4;
  const float scale = 0.125f;                  // 1/sqrt(64)

  // load Q block
  {
    int row = tid >> 1, coff = (tid & 1) * 32;
    const uint4* g = (const uint4*)(Q + ((size_t)(b * SS + q0 + row)) * DD + h * DHH + coff);
    uint4 v0 = g[0], v1 = g[1], v2 = g[2], v3 = g[3];
    *(uint4*)(Qs + row * 64 + coff)      = v0;
    *(uint4*)(Qs + row * 64 + coff + 8)  = v1;
    *(uint4*)(Qs + row * 64 + coff + 16) = v2;
    *(uint4*)(Qs + row * 64 + coff + 24) = v3;
  }
  __syncthreads();

  // persistent Q A-fragments (dh split into 2 K=32 chunks)
  Frag aq[2];
  #pragma unroll
  for (int c = 0; c < 2; ++c) {
    int row = w * 16 + lh;
    aq[c].q[0] = *(const uint4*)(Qs + row * 64 + c * 32 + hb * 8);
    aq[c].q[1] = *(const uint4*)(Qs + row * 64 + c * 32 + hb * 8 + 16);
  }

  v8f zero = {};
  v8f o[4] = {zero, zero, zero, zero};
  float mrow[8], lrow[8];
  #pragma unroll
  for (int r = 0; r < 8; ++r) { mrow[r] = -1e30f; lrow[r] = 0.f; }

  for (int k0 = 0; k0 < SS; k0 += 64) {
    // K/V tile global -> regs
    int row = tid >> 1, coff = (tid & 1) * 32;
    const uint4* gk = (const uint4*)(Kp + ((size_t)(b * SS + k0 + row)) * DD + h * DHH + coff);
    uint4 kv0 = gk[0], kv1 = gk[1], kv2 = gk[2], kv3 = gk[3];
    const uint4* gv = (const uint4*)(V + ((size_t)(b * SS + k0 + row)) * DD + h * DHH + coff);
    unsigned short tv[32];
    *(uint4*)(tv)      = gv[0];
    *(uint4*)(tv + 8)  = gv[1];
    *(uint4*)(tv + 16) = gv[2];
    *(uint4*)(tv + 24) = gv[3];
    __syncthreads();                       // prior iteration finished reading LDS
    *(uint4*)(Ks + row * 64 + coff)      = kv0;
    *(uint4*)(Ks + row * 64 + coff + 8)  = kv1;
    *(uint4*)(Ks + row * 64 + coff + 16) = kv2;
    *(uint4*)(Ks + row * 64 + coff + 24) = kv3;
    #pragma unroll
    for (int j = 0; j < 32; ++j) Vt[(coff + j) * 64 + row] = tv[j];
    __syncthreads();

    // scores: Q (16x64) x K^T (64x64) -> 4 tiles of 16x16, 2 chained WMMAs each
    v8f st[4];
    #pragma unroll
    for (int j = 0; j < 4; ++j) {
      Frag b0, b1;
      int col = j * 16 + lh;               // key index (B-frag column)
      b0.q[0] = *(const uint4*)(Ks + col * 64 + hb * 8);
      b0.q[1] = *(const uint4*)(Ks + col * 64 + hb * 8 + 16);
      b1.q[0] = *(const uint4*)(Ks + col * 64 + 32 + hb * 8);
      b1.q[1] = *(const uint4*)(Ks + col * 64 + 32 + hb * 8 + 16);
      v8f s = __builtin_amdgcn_wmma_f32_16x16x32_bf16(false, aq[0].v, false, b0.v,
                                                      (short)0, zero, false, false);
      s = __builtin_amdgcn_wmma_f32_16x16x32_bf16(false, aq[1].v, false, b1.v,
                                                  (short)0, s, false, false);
      st[j] = s;
    }

    // online softmax (row lives across one 16-lane half)
    #pragma unroll
    for (int r = 0; r < 8; ++r) {
      float s0 = st[0][r] * scale, s1 = st[1][r] * scale;
      float s2 = st[2][r] * scale, s3 = st[3][r] * scale;
      float mx = hmax16(fmaxf(fmaxf(s0, s1), fmaxf(s2, s3)));
      float mnew = fmaxf(mrow[r], mx);
      float fct  = __expf(mrow[r] - mnew);
      float p0 = __expf(s0 - mnew), p1 = __expf(s1 - mnew);
      float p2 = __expf(s2 - mnew), p3 = __expf(s3 - mnew);
      float rs = hsum16(p0 + p1 + p2 + p3);
      lrow[r] = lrow[r] * fct + rs;
      mrow[r] = mnew;
      o[0][r] *= fct; o[1][r] *= fct; o[2][r] *= fct; o[3][r] *= fct;
      int m = r + hb * 8;
      unsigned short* pw = Ps + w * 1024 + m * 64 + lh;
      pw[0]  = f32_to_bf16(p0);
      pw[16] = f32_to_bf16(p1);
      pw[32] = f32_to_bf16(p2);
      pw[48] = f32_to_bf16(p3);
    }

    // P (16x64) x V (64x64): re-read P in A-fragment layout (same-wave LDS, in order)
    Frag ap0, ap1;
    {
      const unsigned short* pb = Ps + w * 1024 + lh * 64;
      ap0.q[0] = *(const uint4*)(pb + hb * 8);
      ap0.q[1] = *(const uint4*)(pb + hb * 8 + 16);
      ap1.q[0] = *(const uint4*)(pb + 32 + hb * 8);
      ap1.q[1] = *(const uint4*)(pb + 32 + hb * 8 + 16);
    }
    #pragma unroll
    for (int jt = 0; jt < 4; ++jt) {
      Frag bv0, bv1;
      int col = jt * 16 + lh;              // dh index (B-frag column in Vt)
      bv0.q[0] = *(const uint4*)(Vt + col * 64 + hb * 8);
      bv0.q[1] = *(const uint4*)(Vt + col * 64 + hb * 8 + 16);
      bv1.q[0] = *(const uint4*)(Vt + col * 64 + 32 + hb * 8);
      bv1.q[1] = *(const uint4*)(Vt + col * 64 + 32 + hb * 8 + 16);
      o[jt] = __builtin_amdgcn_wmma_f32_16x16x32_bf16(false, ap0.v, false, bv0.v,
                                                      (short)0, o[jt], false, false);
      o[jt] = __builtin_amdgcn_wmma_f32_16x16x32_bf16(false, ap1.v, false, bv1.v,
                                                      (short)0, o[jt], false, false);
    }
  }

  // normalize + store
  #pragma unroll
  for (int jt = 0; jt < 4; ++jt)
    #pragma unroll
    for (int r = 0; r < 8; ++r) {
      int m = r + hb * 8;
      float val = o[jt][r] / lrow[r];
      Op[((size_t)(b * SS + q0 + w * 16 + m)) * DD + h * DHH + jt * 16 + lh] =
          f32_to_bf16(val);
    }
}

// ---------------------------------------------------------------- launcher
extern "C" void kernel_launch(void* const* d_in, const int* in_sizes, int n_in,
                              void* d_out, int out_size, void* d_ws, size_t ws_size,
                              hipStream_t stream) {
  (void)in_sizes; (void)n_in; (void)out_size; (void)ws_size;
  const float* x  = (const float*)d_in[0];
  const float* wq = (const float*)d_in[1];
  const float* wk = (const float*)d_in[2];
  const float* wv = (const float*)d_in[3];
  const float* wo = (const float*)d_in[4];
  const float* g1 = (const float*)d_in[5];
  const float* g2 = (const float*)d_in[6];
  const float* w1 = (const float*)d_in[7];
  const float* w2 = (const float*)d_in[8];
  const float* w3 = (const float*)d_in[9];
  float* out = (float*)d_out;

  char* ws = (char*)d_ws;
  size_t off = 0;
  auto carve = [&](size_t bytes) -> void* {
    void* p = ws + off;
    off = (off + bytes + 255) & ~(size_t)255;
    return p;
  };
  const size_t nDD  = (size_t)DD * DD;     // 1M
  const size_t nDF  = (size_t)DD * FFF;    // 4M
  const size_t nAct = (size_t)MROWS * DD;  // 4M
  const size_t nH   = (size_t)MROWS * FFF; // 16M

  unsigned short* wq_bf  = (unsigned short*)carve(nDD * 2);
  unsigned short* wk_bf  = (unsigned short*)carve(nDD * 2);
  unsigned short* wv_bf  = (unsigned short*)carve(nDD * 2);
  unsigned short* wo_bf  = (unsigned short*)carve(nDD * 2);
  unsigned short* w1_bf  = (unsigned short*)carve(nDF * 2);
  unsigned short* w2_bf  = (unsigned short*)carve(nDF * 2);
  unsigned short* w3_bf  = (unsigned short*)carve(nDF * 2);
  unsigned short* xn_bf  = (unsigned short*)carve(nAct * 2);
  unsigned short* q_bf   = (unsigned short*)carve(nAct * 2);
  unsigned short* k_bf   = (unsigned short*)carve(nAct * 2);
  unsigned short* v_bf   = (unsigned short*)carve(nAct * 2);
  unsigned short* at_bf  = (unsigned short*)carve(nAct * 2);
  float*          xmid   = (float*)carve(nAct * 4);
  unsigned short* xn2_bf = (unsigned short*)carve(nAct * 2);
  unsigned short* h1_bf  = (unsigned short*)carve(nH * 2);
  unsigned short* h_bf   = (unsigned short*)carve(nH * 2);

  // 1. weights -> bf16
  cvt_f32_bf16<<<(unsigned)((nDD + 255) / 256), 256, 0, stream>>>(wq, wq_bf, nDD);
  cvt_f32_bf16<<<(unsigned)((nDD + 255) / 256), 256, 0, stream>>>(wk, wk_bf, nDD);
  cvt_f32_bf16<<<(unsigned)((nDD + 255) / 256), 256, 0, stream>>>(wv, wv_bf, nDD);
  cvt_f32_bf16<<<(unsigned)((nDD + 255) / 256), 256, 0, stream>>>(wo, wo_bf, nDD);
  cvt_f32_bf16<<<(unsigned)((nDF + 255) / 256), 256, 0, stream>>>(w1, w1_bf, nDF);
  cvt_f32_bf16<<<(unsigned)((nDF + 255) / 256), 256, 0, stream>>>(w2, w2_bf, nDF);
  cvt_f32_bf16<<<(unsigned)((nDF + 255) / 256), 256, 0, stream>>>(w3, w3_bf, nDF);

  // 2. xn = rmsnorm(x, g1)
  rmsnorm_kernel<<<MROWS, 256, 0, stream>>>(x, g1, xn_bf, DD);

  // 3. q/k/v projections (bf16 WMMA GEMMs, TDM-fed A tiles)
  dim3 gqkv(DD / 128, MROWS / 128);
  gemm_bf16_128x128<0><<<gqkv, 256, 0, stream>>>(xn_bf, wq_bf, q_bf, nullptr, nullptr, MROWS, DD, DD);
  gemm_bf16_128x128<0><<<gqkv, 256, 0, stream>>>(xn_bf, wk_bf, k_bf, nullptr, nullptr, MROWS, DD, DD);
  gemm_bf16_128x128<0><<<gqkv, 256, 0, stream>>>(xn_bf, wv_bf, v_bf, nullptr, nullptr, MROWS, DD, DD);

  // 4. RoPE on q and k
  const unsigned ropeBlocks = (unsigned)(((size_t)BB * SS * HH * 32) / 256);
  rope_kernel<<<ropeBlocks, 256, 0, stream>>>(q_bf);
  rope_kernel<<<ropeBlocks, 256, 0, stream>>>(k_bf);

  // 5. flash attention
  flash_attn_kernel<<<dim3(SS / 64, HH, BB), 128, 0, stream>>>(q_bf, k_bf, v_bf, at_bf);

  // 6. xmid = x + attn @ wo
  gemm_bf16_128x128<1><<<gqkv, 256, 0, stream>>>(at_bf, wo_bf, xmid, x, nullptr, MROWS, DD, DD);

  // 7. xn2 = rmsnorm(xmid, g2)
  rmsnorm_kernel<<<MROWS, 256, 0, stream>>>(xmid, g2, xn2_bf, DD);

  // 8. FFN: h1 = xn2@w1 ; h = silu(h1) * (xn2@w2) ; out = xmid + h@w3
  dim3 gffn(FFF / 128, MROWS / 128);
  gemm_bf16_128x128<0><<<gffn, 256, 0, stream>>>(xn2_bf, w1_bf, h1_bf, nullptr, nullptr, MROWS, FFF, DD);
  gemm_bf16_128x128<2><<<gffn, 256, 0, stream>>>(xn2_bf, w2_bf, h_bf, nullptr, h1_bf, MROWS, FFF, DD);
  dim3 gout(DD / 128, MROWS / 128);
  gemm_bf16_128x128<1><<<gout, 256, 0, stream>>>(h_bf, w3_bf, out, xmid, nullptr, MROWS, DD, FFF);
}
